// ResGCN_BN_5583457484893
// MI455X (gfx1250) — matrix operand order repaired and checked by reading, not compiled, not run on hardware
//
#include <hip/hip_runtime.h>

// ---------------------------------------------------------------------------
// ResGCN on MI455X (gfx1250): bf16 WMMA GEMM pipeline.
//  - adj cast to bf16 once -> 134MB, L2-resident (192MB L2) for 6 adj-GEMMs.
//  - one generic batched GEMM: C = A(bf16,row) @ B (given as B^T bf16 rows)
//    with fused bias / skip / BN / ReLU epilogue, writing fp32 and/or bf16
//    row-major and/or bf16 transposed (feeds next GEMM's B^T directly).
//  - v_wmma_f32_16x16x32_bf16, wave32, 4 waves/block, 64x64 block tile,
//    2x2 register blocking per wave (4 WMMAs per 8 b128 loads).
// ---------------------------------------------------------------------------

typedef __attribute__((ext_vector_type(16))) __bf16 v16bf;
typedef __attribute__((ext_vector_type(8)))  __bf16 v8bf;
typedef __attribute__((ext_vector_type(8)))  float  v8f;

// ------------------------------ small kernels ------------------------------

__global__ __launch_bounds__(256) void cast_bf16_kernel(
    const float* __restrict__ in, __bf16* __restrict__ out, long n)
{
    long i = (long)blockIdx.x * blockDim.x + threadIdx.x;
    if (i < n) out[i] = (__bf16)in[i];
}

// out[b][c][r] = in[b][r][c]  (fp32 -> bf16 transpose); grid.z = batch
__global__ __launch_bounds__(256) void transpose_cast_kernel(
    const float* __restrict__ in, __bf16* __restrict__ out,
    int rows, int cols, long sIn, long sOut)
{
    long idx = (long)blockIdx.x * blockDim.x + threadIdx.x;
    long total = (long)rows * cols;
    if (idx >= total) return;
    int b = blockIdx.z;
    int r = (int)(idx / cols), c = (int)(idx % cols);
    out[(long)b * sOut + (long)c * rows + r] = (__bf16)in[(long)b * sIn + idx];
}

__global__ __launch_bounds__(128) void bn_prep_kernel(
    const float* __restrict__ g, const float* __restrict__ be,
    const float* __restrict__ rm, const float* __restrict__ rv,
    float* __restrict__ scale, float* __restrict__ shift, int n)
{
    int i = threadIdx.x;
    if (i < n) {
        float s = g[i] * rsqrtf(rv[i] + 1e-5f);
        scale[i] = s;
        shift[i] = be[i] - rm[i] * s;
    }
}

// out = x[:, :, 0:2] @ W[2,H] (+bias): trivial K=2 contraction on VALU.
__global__ __launch_bounds__(256) void lin_k2_kernel(
    const float* __restrict__ x, const float* __restrict__ W,
    const float* __restrict__ bias,
    float* __restrict__ outF, __bf16* __restrict__ outT, int Nr, int Hc)
{
    long idx = (long)blockIdx.x * blockDim.x + threadIdx.x;
    long total = (long)Nr * Hc;
    if (idx >= total) return;
    int b = blockIdx.z;
    int n = (int)(idx / Hc), h = (int)(idx % Hc);
    const float* xp = x + ((long)b * Nr + n) * 2;
    float v = xp[0] * W[h] + xp[1] * W[Hc + h];
    if (bias) v += bias[h];
    if (outF) outF[(long)b * total + idx] = v;
    if (outT) outT[(long)b * total + (long)h * Nr + n] = (__bf16)v;
}

// out = a[:, :, 0:1] @ W[1,H] (+bias): rank-1 outer product on VALU.
__global__ __launch_bounds__(256) void outer_k1_kernel(
    const float* __restrict__ a, const float* __restrict__ W,
    const float* __restrict__ bias,
    float* __restrict__ outF, __bf16* __restrict__ outT, int Mr, int Hc)
{
    long idx = (long)blockIdx.x * blockDim.x + threadIdx.x;
    long total = (long)Mr * Hc;
    if (idx >= total) return;
    int b = blockIdx.z;
    int m = (int)(idx / Hc), h = (int)(idx % Hc);
    float v = a[(long)b * Mr + m] * W[h];
    if (bias) v += bias[h];
    if (outF) outF[(long)b * total + idx] = v;
    if (outT) outT[(long)b * total + (long)h * Mr + m] = (__bf16)v;
}

// out_bf16 = relu(in * scale[h] + shift[h])
__global__ __launch_bounds__(256) void bn_relu_cast_kernel(
    const float* __restrict__ in, const float* __restrict__ scale,
    const float* __restrict__ shift, __bf16* __restrict__ out, int Hc, long n)
{
    long i = (long)blockIdx.x * blockDim.x + threadIdx.x;
    if (i >= n) return;
    int h = (int)(i % Hc);
    float v = in[i] * scale[h] + shift[h];
    out[i] = (__bf16)fmaxf(v, 0.f);
}

// d_out[0:n) = sigmoid(raw), d_out[n:2n) = raw
__global__ __launch_bounds__(256) void sigmoid_pair_kernel(
    const float* __restrict__ in, float* __restrict__ out, long n)
{
    long i = (long)blockIdx.x * blockDim.x + threadIdx.x;
    if (i >= n) return;
    float v = in[i];
    out[i] = 1.f / (1.f + expf(-v));
    out[n + i] = v;
}

// ------------------------------ WMMA GEMM ----------------------------------
// C[b] = A[b] @ B[b] where B is supplied transposed (BT rows = B columns).
// Block: 128 threads = 4 waves in 2x2; block tile 64(M) x 64(N).
// Each wave: 32x32 output = 2x2 WMMA tiles sharing 2 A + 2 B fragments.
__global__ __launch_bounds__(128) void gemm_bf16_wmma(
    const __bf16* __restrict__ A, long sA, int lda,
    const __bf16* __restrict__ BT, long sBT, int ldbt,
    const float* __restrict__ bias,
    const float* __restrict__ skip, long sSkip, int ldskip,
    const float* __restrict__ bnscale, const float* __restrict__ bnshift,
    int relu,
    float* __restrict__ Cf, long sCf, int ldcf,
    __bf16* __restrict__ Cb, long sCb, int ldcb,
    __bf16* __restrict__ CbT, long sCbT, int ldcbt,
    int Mdim, int Ndim, int Kdim)
{
    const int lane = threadIdx.x & 31;
    const int wid  = threadIdx.x >> 5;                 // 0..3
    const int tileM = blockIdx.y * 64 + (wid >> 1) * 32;
    const int tileN = blockIdx.x * 64 + (wid & 1) * 32;
    if (tileM >= Mdim || tileN >= Ndim) return;        // uniform per-wave exit
    const int b = blockIdx.z;
    const int l15 = lane & 15;
    const int khalfA = (lane >> 4) << 3;   // A frag: lanes 16-31 start at K+8
    const int khalfB = (lane >> 4) << 3;   // B frag: lanes 16-31 start at K+8

    const __bf16* ap0 = A  + (long)b * sA  + (long)(tileM + l15) * lda  + khalfA;
    const __bf16* ap1 = ap0 + (long)16 * lda;
    const __bf16* bp0 = BT + (long)b * sBT + (long)(tileN + l15) * ldbt + khalfB;
    const __bf16* bp1 = bp0 + (long)16 * ldbt;

    v8f acc[2][2];
#pragma unroll
    for (int tm = 0; tm < 2; ++tm)
#pragma unroll
        for (int tn = 0; tn < 2; ++tn)
            acc[tm][tn] = (v8f){0.f, 0.f, 0.f, 0.f, 0.f, 0.f, 0.f, 0.f};

    for (int k = 0; k < Kdim; k += 32) {
        union { v16bf v; v8bf h[2]; } a0, a1;
        a0.h[0] = *(const v8bf*)(ap0 + k);        // K = k+khalf   .. +7
        a0.h[1] = *(const v8bf*)(ap0 + k + 16);   // K = k+16+khalf.. +7
        a1.h[0] = *(const v8bf*)(ap1 + k);
        a1.h[1] = *(const v8bf*)(ap1 + k + 16);
        union { v16bf v; v8bf h[2]; } b0u, b1u;
        b0u.h[0] = *(const v8bf*)(bp0 + k);       // K = k+khalfB  .. +7
        b0u.h[1] = *(const v8bf*)(bp0 + k + 16);  // K = k+16+khalfB.. +7
        b1u.h[0] = *(const v8bf*)(bp1 + k);
        b1u.h[1] = *(const v8bf*)(bp1 + k + 16);
        v16bf b0 = b0u.v;
        v16bf b1 = b1u.v;
        acc[0][0] = __builtin_amdgcn_wmma_f32_16x16x32_bf16(
            false, a0.v, false, b0, (short)0, acc[0][0], false, false);
        acc[0][1] = __builtin_amdgcn_wmma_f32_16x16x32_bf16(
            false, a0.v, false, b1, (short)0, acc[0][1], false, false);
        acc[1][0] = __builtin_amdgcn_wmma_f32_16x16x32_bf16(
            false, a1.v, false, b0, (short)0, acc[1][0], false, false);
        acc[1][1] = __builtin_amdgcn_wmma_f32_16x16x32_bf16(
            false, a1.v, false, b1, (short)0, acc[1][1], false, false);
    }

    // epilogue: D[m][n], m = base + r + 8*(lane>>4), n = base + (lane&15)
    const int mhalf = (lane >> 4) << 3;
#pragma unroll
    for (int tm = 0; tm < 2; ++tm) {
#pragma unroll
        for (int tn = 0; tn < 2; ++tn) {
            const int nn = tileN + tn * 16 + l15;
            const float bsv = bias    ? bias[nn]    : 0.f;
            const float sc  = bnscale ? bnscale[nn] : 1.f;
            const float sh  = bnshift ? bnshift[nn] : 0.f;
            const int mbase = tileM + tm * 16 + mhalf;
#pragma unroll
            for (int r = 0; r < 8; ++r) {
                const int mm = mbase + r;
                float v = acc[tm][tn][r] + bsv;
                if (skip)    v += skip[(long)b * sSkip + (long)mm * ldskip + nn];
                if (bnscale) v = v * sc + sh;
                if (relu)    v = fmaxf(v, 0.f);
                if (Cf)  Cf [(long)b * sCf  + (long)mm * ldcf  + nn] = v;
                if (Cb)  Cb [(long)b * sCb  + (long)mm * ldcb  + nn] = (__bf16)v;
                if (CbT) CbT[(long)b * sCbT + (long)nn * ldcbt + mm] = (__bf16)v;
            }
        }
    }
}

// ------------------------------ host side ----------------------------------

static void launch_gemm(hipStream_t st,
    const __bf16* A, long sA, int lda,
    const __bf16* BT, long sBT, int ldbt,
    const float* bias,
    const float* skip, long sSkip, int ldskip,
    const float* scale, const float* shift, int relu,
    float* Cf, long sCf, int ldcf,
    __bf16* Cb, long sCb, int ldcb,
    __bf16* CbT, long sCbT, int ldcbt,
    int Md, int Nd, int Kd, int batches)
{
    dim3 g((unsigned)((Nd + 63) / 64), (unsigned)((Md + 63) / 64), (unsigned)batches);
    gemm_bf16_wmma<<<g, 128, 0, st>>>(A, sA, lda, BT, sBT, ldbt, bias,
                                      skip, sSkip, ldskip, scale, shift, relu,
                                      Cf, sCf, ldcf, Cb, sCb, ldcb,
                                      CbT, sCbT, ldcbt, Md, Nd, Kd);
}

extern "C" void kernel_launch(void* const* d_in, const int* in_sizes, int n_in,
                              void* d_out, int out_size, void* d_ws, size_t ws_size,
                              hipStream_t stream)
{
    (void)in_sizes; (void)n_in; (void)out_size; (void)ws_size;
    const int B = 4, N = 4096, M = 256, H = 128, C = 64;

    const float* x    = (const float*)d_in[0];   // [B,N,2]
    const float* att  = (const float*)d_in[1];   // [B,M,1]
    const float* adj  = (const float*)d_in[2];   // [B,N,N]
    const float* feat = (const float*)d_in[3];   // [B,N,M]

    struct Lin { const float *W, *b; };
    struct Bn  { const float *g, *be, *rm, *rv; };
    int pi = 4;
    auto nextLin = [&]() { Lin l{(const float*)d_in[pi], (const float*)d_in[pi + 1]}; pi += 2; return l; };
    auto nextBn  = [&]() { Bn  b{(const float*)d_in[pi], (const float*)d_in[pi + 1],
                                 (const float*)d_in[pi + 2], (const float*)d_in[pi + 3]}; pi += 4; return b; };
    // params dict insertion order (each lin = {W,b}, each bn = {g,be,rm,rv})
    Lin gc1 = nextLin(), sl1s = nextLin(), fc1 = nextLin(), fd1 = nextLin(),
        sl1fa = nextLin(), cnd1 = nextLin(), sl1f = nextLin(),
        gc2 = nextLin(), sl2s = nextLin(), fc2 = nextLin(), fd2 = nextLin(),
        sl2fa = nextLin(), cnd2 = nextLin(), sl2f = nextLin(),
        gc3 = nextLin(), sl3s = nextLin(), fc3 = nextLin(), fd3 = nextLin(),
        sl3fa = nextLin(), cnd3 = nextLin();
    Bn bns[6] = {nextBn(), nextBn(), nextBn(), nextBn(), nextBn(), nextBn()};
    // 0:bn1 1:bn12 2:bn13 3:bn2 4:bn21 5:bn23

    // ---- workspace bump allocator (256B aligned) ----
    char* wsp = (char*)d_ws;
    size_t off = 0;
    auto alloc = [&](size_t bytes) -> void* {
        void* p = wsp + off;
        off = (off + bytes + 255) & ~(size_t)255;
        return p;
    };
    __bf16* adjb   = (__bf16*)alloc((size_t)B * N * N * 2);   // 134 MB, L2-resident
    __bf16* featb  = (__bf16*)alloc((size_t)B * N * M * 2);
    __bf16* featTb = (__bf16*)alloc((size_t)B * M * N * 2);   // ft = feat^T
    __bf16* R1T    = (__bf16*)alloc((size_t)B * H * N * 2);
    __bf16* R3T    = (__bf16*)alloc((size_t)B * H * N * 2);
    __bf16* PT     = (__bf16*)alloc((size_t)B * H * M * 2);
    float*  S1     = (float*) alloc((size_t)B * N * H * 4);
    float*  S3     = (float*) alloc((size_t)B * N * H * 4);
    float*  S2     = (float*) alloc((size_t)B * M * H * 4);
    float*  X3raw  = (float*) alloc((size_t)B * N * H * 4);
    __bf16* cat    = (__bf16*)alloc((size_t)B * N * 3 * H * 2);
    __bf16* hb     = (__bf16*)alloc((size_t)B * N * H * 2);
    __bf16* hT     = (__bf16*)alloc((size_t)B * H * N * 2);
    __bf16* x2b    = (__bf16*)alloc((size_t)B * M * H * 2);
    __bf16* x3b    = (__bf16*)alloc((size_t)B * N * H * 2);
    float*  outraw = (float*) alloc((size_t)B * N * C * 4);
    float*  bnbuf  = (float*) alloc((size_t)12 * H * 4);

    auto wTt = [&](const float* W, int K, int D) -> __bf16* {  // [K,D] -> bf16 [D,K]
        __bf16* p = (__bf16*)alloc((size_t)K * D * 2);
        transpose_cast_kernel<<<dim3((unsigned)((K * D + 255) / 256), 1, 1), 256, 0, stream>>>(
            W, p, K, D, 0, 0);
        return p;
    };

    // ---- stage 0: casts / transposes / BN folding ----
    long adjE = (long)B * N * N, featE = (long)B * N * M;
    cast_bf16_kernel<<<dim3((unsigned)((adjE + 255) / 256)), 256, 0, stream>>>(adj, adjb, adjE);
    cast_bf16_kernel<<<dim3((unsigned)((featE + 255) / 256)), 256, 0, stream>>>(feat, featb, featE);
    transpose_cast_kernel<<<dim3((unsigned)((N * M + 255) / 256), 1, B), 256, 0, stream>>>(
        feat, featTb, N, M, (long)N * M, (long)M * N);

    __bf16 *Wfd1T = wTt(fd1.W, M, H),  *Wsl1faT = wTt(sl1fa.W, M, H),
           *Wcnd1T = wTt(cnd1.W, 3 * H, H),
           *Wgc2T = wTt(gc2.W, H, H),  *Wsl2sT = wTt(sl2s.W, H, H),
           *Wfc2T = wTt(fc2.W, H, H),  *Wfd2T = wTt(fd2.W, H, H),
           *Wsl2faT = wTt(sl2fa.W, H, H), *Wsl2fT = wTt(sl2f.W, H, H),
           *Wcnd2T = wTt(cnd2.W, 3 * H, H),
           *Wgc3T = wTt(gc3.W, H, C),  *Wsl3sT = wTt(sl3s.W, H, C),
           *Wfc3T = wTt(fc3.W, H, C),  *Wfd3T = wTt(fd3.W, H, C),
           *Wsl3faT = wTt(sl3fa.W, H, C), *Wcnd3T = wTt(cnd3.W, 3 * C, C);

    for (int j = 0; j < 6; ++j)
        bn_prep_kernel<<<1, 128, 0, stream>>>(bns[j].g, bns[j].be, bns[j].rm, bns[j].rv,
                                              bnbuf + j * 2 * H, bnbuf + j * 2 * H + H, H);
    auto SC = [&](int j) { return bnbuf + j * 2 * H; };
    auto SH = [&](int j) { return bnbuf + j * 2 * H + H; };

    const dim3 gNH((unsigned)(((long)N * H + 255) / 256), 1, (unsigned)B);
    const dim3 gMH((unsigned)(((long)M * H + 255) / 256), 1, (unsigned)B);
    const long sNH = (long)N * H, sMH = (long)M * H, sHN = (long)H * N,
               sHM = (long)H * M, sNN = (long)N * N, sNM = (long)N * M,
               sCat = (long)N * 3 * H, sNC = (long)N * C, sCN = (long)C * N,
               sCM = (long)C * M, sCat3 = (long)N * 3 * C;

    // =================== layer 1 ===================
    lin_k2_kernel<<<gNH, 256, 0, stream>>>(x, gc1.W, nullptr, nullptr, R1T, N, H); // R1^T
    lin_k2_kernel<<<gNH, 256, 0, stream>>>(x, sl1s.W, sl1s.b, S1, nullptr, N, H);  // S1
    outer_k1_kernel<<<gMH, 256, 0, stream>>>(att, fc1.W, nullptr, nullptr, PT, M, H);   // P1^T
    outer_k1_kernel<<<gMH, 256, 0, stream>>>(att, sl1f.W, sl1f.b, S2, nullptr, M, H);   // S2
    // R3^T = (feat @ Wfd1)^T ; S3 = feat @ Wsl1fa + b
    launch_gemm(stream, featb, sNM, M, Wfd1T, 0, M, nullptr, nullptr, 0, 0, nullptr, nullptr, 0,
                nullptr, 0, 0, nullptr, 0, 0, R3T, sHN, N, N, H, M, B);
    launch_gemm(stream, featb, sNM, M, Wsl1faT, 0, M, sl1fa.b, nullptr, 0, 0, nullptr, nullptr, 0,
                S3, sNH, H, nullptr, 0, 0, nullptr, 0, 0, N, H, M, B);
    // X1 = adj@R1 + b_gc1 + S1 -> cat[:,0:H]
    launch_gemm(stream, adjb, sNN, N, R1T, sHN, N, gc1.b, S1, sNH, H, nullptr, nullptr, 0,
                nullptr, 0, 0, cat, sCat, 3 * H, nullptr, 0, 0, N, H, N, B);
    // X2 = feat@P1 + b_fc1 -> cat[:,H:2H]
    launch_gemm(stream, featb, sNM, M, PT, sHM, M, fc1.b, nullptr, 0, 0, nullptr, nullptr, 0,
                nullptr, 0, 0, cat + H, sCat, 3 * H, nullptr, 0, 0, N, H, M, B);
    // X3 = adj@R3 + b_fd1 + S3 -> X3raw + cat[:,2H:3H]
    launch_gemm(stream, adjb, sNN, N, R3T, sHN, N, fd1.b, S3, sNH, H, nullptr, nullptr, 0,
                X3raw, sNH, H, cat + 2 * H, sCat, 3 * H, nullptr, 0, 0, N, H, N, B);
    // h = relu(bn1(cat @ Wcnd1 + b)) -> hb, hT
    launch_gemm(stream, cat, sCat, 3 * H, Wcnd1T, 0, 3 * H, cnd1.b, nullptr, 0, 0, SC(0), SH(0), 1,
                nullptr, 0, 0, hb, sNH, H, hT, sHN, N, N, H, 3 * H, B);
    // x2 = relu(bn12(ft@h + S2)) -> x2b
    launch_gemm(stream, featTb, (long)M * N, N, hT, sHN, N, nullptr, S2, sMH, H, SC(1), SH(1), 1,
                nullptr, 0, 0, x2b, sMH, H, nullptr, 0, 0, M, H, N, B);
    // x3 = relu(bn13(X3raw)) -> x3b
    bn_relu_cast_kernel<<<dim3((unsigned)(((long)B * N * H + 255) / 256)), 256, 0, stream>>>(
        X3raw, SC(2), SH(2), x3b, H, (long)B * N * H);

    // =================== layer 2 ===================
    // S2' = y2 @ Wsl2f + b  (consumes old x2 before overwrite)
    launch_gemm(stream, x2b, sMH, H, Wsl2fT, 0, H, sl2f.b, nullptr, 0, 0, nullptr, nullptr, 0,
                S2, sMH, H, nullptr, 0, 0, nullptr, 0, 0, M, H, H, B);
    // P2^T = (x2 @ Wfc2)^T
    launch_gemm(stream, x2b, sMH, H, Wfc2T, 0, H, nullptr, nullptr, 0, 0, nullptr, nullptr, 0,
                nullptr, 0, 0, nullptr, 0, 0, PT, sHM, M, M, H, H, B);
    // R1^T = (h @ Wgc2)^T ; S1 = h @ Wsl2s + b
    launch_gemm(stream, hb, sNH, H, Wgc2T, 0, H, nullptr, nullptr, 0, 0, nullptr, nullptr, 0,
                nullptr, 0, 0, nullptr, 0, 0, R1T, sHN, N, N, H, H, B);
    launch_gemm(stream, hb, sNH, H, Wsl2sT, 0, H, sl2s.b, nullptr, 0, 0, nullptr, nullptr, 0,
                S1, sNH, H, nullptr, 0, 0, nullptr, 0, 0, N, H, H, B);
    // R3^T = (x3 @ Wfd2)^T ; S3 = x3 @ Wsl2fa + b
    launch_gemm(stream, x3b, sNH, H, Wfd2T, 0, H, nullptr, nullptr, 0, 0, nullptr, nullptr, 0,
                nullptr, 0, 0, nullptr, 0, 0, R3T, sHN, N, N, H, H, B);
    launch_gemm(stream, x3b, sNH, H, Wsl2faT, 0, H, sl2fa.b, nullptr, 0, 0, nullptr, nullptr, 0,
                S3, sNH, H, nullptr, 0, 0, nullptr, 0, 0, N, H, H, B);
    // X1 / X2 / X3 -> cat
    launch_gemm(stream, adjb, sNN, N, R1T, sHN, N, gc2.b, S1, sNH, H, nullptr, nullptr, 0,
                nullptr, 0, 0, cat, sCat, 3 * H, nullptr, 0, 0, N, H, N, B);
    launch_gemm(stream, featb, sNM, M, PT, sHM, M, fc2.b, nullptr, 0, 0, nullptr, nullptr, 0,
                nullptr, 0, 0, cat + H, sCat, 3 * H, nullptr, 0, 0, N, H, M, B);
    launch_gemm(stream, adjb, sNN, N, R3T, sHN, N, fd2.b, S3, sNH, H, nullptr, nullptr, 0,
                X3raw, sNH, H, cat + 2 * H, sCat, 3 * H, nullptr, 0, 0, N, H, N, B);
    // h2 = relu(bn2(cat @ Wcnd2 + b)) -> hb, hT
    launch_gemm(stream, cat, sCat, 3 * H, Wcnd2T, 0, 3 * H, cnd2.b, nullptr, 0, 0, SC(3), SH(3), 1,
                nullptr, 0, 0, hb, sNH, H, hT, sHN, N, N, H, 3 * H, B);
    // x2 = relu(bn21(ft@h2 + S2')) -> x2b
    launch_gemm(stream, featTb, (long)M * N, N, hT, sHN, N, nullptr, S2, sMH, H, SC(4), SH(4), 1,
                nullptr, 0, 0, x2b, sMH, H, nullptr, 0, 0, M, H, N, B);
    // x3 = relu(bn23(X3raw)) -> x3b
    bn_relu_cast_kernel<<<dim3((unsigned)(((long)B * N * H + 255) / 256)), 256, 0, stream>>>(
        X3raw, SC(5), SH(5), x3b, H, (long)B * N * H);

    // =================== layer 3 (width C=64) ===================
    launch_gemm(stream, hb, sNH, H, Wgc3T, 0, H, nullptr, nullptr, 0, 0, nullptr, nullptr, 0,
                nullptr, 0, 0, nullptr, 0, 0, R1T, sCN, N, N, C, H, B);          // R1^T [B,C,N]
    launch_gemm(stream, hb, sNH, H, Wsl3sT, 0, H, sl3s.b, nullptr, 0, 0, nullptr, nullptr, 0,
                S1, sNC, C, nullptr, 0, 0, nullptr, 0, 0, N, C, H, B);           // S1c
    launch_gemm(stream, x2b, sMH, H, Wfc3T, 0, H, nullptr, nullptr, 0, 0, nullptr, nullptr, 0,
                nullptr, 0, 0, nullptr, 0, 0, PT, sCM, M, M, C, H, B);           // P3^T [B,C,M]
    launch_gemm(stream, x3b, sNH, H, Wfd3T, 0, H, nullptr, nullptr, 0, 0, nullptr, nullptr, 0,
                nullptr, 0, 0, nullptr, 0, 0, R3T, sCN, N, N, C, H, B);          // R3^T
    launch_gemm(stream, x3b, sNH, H, Wsl3faT, 0, H, sl3fa.b, nullptr, 0, 0, nullptr, nullptr, 0,
                S3, sNC, C, nullptr, 0, 0, nullptr, 0, 0, N, C, H, B);           // S3c
    launch_gemm(stream, adjb, sNN, N, R1T, sCN, N, gc3.b, S1, sNC, C, nullptr, nullptr, 0,
                nullptr, 0, 0, cat, sCat3, 3 * C, nullptr, 0, 0, N, C, N, B);    // X1
    launch_gemm(stream, featb, sNM, M, PT, sCM, M, fc3.b, nullptr, 0, 0, nullptr, nullptr, 0,
                nullptr, 0, 0, cat + C, sCat3, 3 * C, nullptr, 0, 0, N, C, M, B); // X2
    launch_gemm(stream, adjb, sNN, N, R3T, sCN, N, fd3.b, S3, sNC, C, nullptr, nullptr, 0,
                nullptr, 0, 0, cat + 2 * C, sCat3, 3 * C, nullptr, 0, 0, N, C, N, B); // X3
    // out = cat3 @ Wcnd3 + b
    launch_gemm(stream, cat, sCat3, 3 * C, Wcnd3T, 0, 3 * C, cnd3.b, nullptr, 0, 0, nullptr, nullptr, 0,
                outraw, sNC, C, nullptr, 0, 0, nullptr, 0, 0, N, C, 3 * C, B);

    long nOut = (long)B * N * C;
    sigmoid_pair_kernel<<<dim3((unsigned)((nOut + 255) / 256)), 256, 0, stream>>>(
        outraw, (float*)d_out, nOut);
}
